// MoELoRA_47871705481666
// MI455X (gfx1250) — matrix operand order repaired and checked by reading, not compiled, not run on hardware
//
#include <hip/hip_runtime.h>

// ---- problem constants ----
#define TOK    16384   // B*S
#define DDIM   2048
#define NEXP   64
#define ERANK  512     // E*R
#define RD     16      // RDIM
#define KSEL   16      // TOPK
#define SCALE_OUT 2.0f // ALPHA/TOPK = 32/16

// ---- GEMM tiling ----
#define BM  128
#define BN  128
#define BK  32
#define LDT 40         // LDS row stride in bf16 (32 + 8 pad) = 80B; TDM pad reproduces this

typedef __attribute__((ext_vector_type(8)))  __bf16 v8bf;
typedef __attribute__((ext_vector_type(16))) __bf16 v16bf;
typedef __attribute__((ext_vector_type(8)))  float  v8f;
typedef __attribute__((ext_vector_type(4)))  unsigned int u32x4;
typedef __attribute__((ext_vector_type(8)))  int i32x8;
typedef __attribute__((ext_vector_type(4)))  int i32x4;

// ------------------------------------------------------------------
// TDM: DMA one 128-row x 32-col bf16 tile (row stride in elements)
// into LDS, padding each 64B row by 16B -> LDS row stride 80B (LDT).
// D# layout per CDNA5 ISA.
// ------------------------------------------------------------------
__device__ inline void tdm_load_tile_bf16(unsigned lds_addr, const void* gsrc,
                                          unsigned long long stride_elems) {
  unsigned long long ga = (unsigned long long)gsrc;
  u32x4 g0;
  g0[0] = 1u;                                     // count=1, user mode
  g0[1] = lds_addr;                               // lds_addr
  g0[2] = (unsigned)ga;                           // global_addr[31:0]
  g0[3] = (unsigned)((ga >> 32) & 0x01ffffffu)    // global_addr[56:32]
        | 0x80000000u;                            // type=2 ("image")
  i32x8 g1;
  g1[0] = (1 << 16)                               // data_size = 2 bytes
        | (1 << 20)                               // pad_enable
        | (3 << 22)                               // pad_interval = 16 DWORDs (64B row)
        | (3 << 25);                              // pad_amount   = 4 DWORDs (16B)
  g1[1] = (BK & 0xffff) << 16;                    // tensor_dim0 = 32
  g1[2] = (BN & 0xffff) << 16;                    // tensor_dim1 = 128 (lo16)
  g1[3] = (BK & 0xffff) << 16;                    // tile_dim0 = 32
  g1[4] = (BN & 0xffff);                          // tile_dim1 = 128, tile_dim2 = 0
  g1[5] = (int)(stride_elems & 0xffffffffu);      // tensor_dim0_stride[31:0]
  g1[6] = (int)((stride_elems >> 32) & 0xffffu);  // stride[47:32]
  g1[7] = 0;
  i32x4 gz = {0, 0, 0, 0};
#if __clang_major__ >= 23
  i32x8 gz8 = {0, 0, 0, 0, 0, 0, 0, 0};
  __builtin_amdgcn_tensor_load_to_lds(g0, g1, gz, gz, gz8, 0);
#else
  __builtin_amdgcn_tensor_load_to_lds(g0, g1, gz, gz, 0);
#endif
}

// ------------------------------------------------------------------
// f32 -> bf16 convert
// ------------------------------------------------------------------
__global__ __launch_bounds__(256) void cvt_bf16(const float* __restrict__ in,
                                                __bf16* __restrict__ out, int n) {
  int i = blockIdx.x * 256 + threadIdx.x;
  if (i < n) out[i] = (__bf16)in[i];
}

// ------------------------------------------------------------------
// Router: logits = (x @ Wr1) @ Wr2 ; softmax ; top-16 gate ; probs
// ------------------------------------------------------------------
__global__ __launch_bounds__(64) void router_kernel(
    const float* __restrict__ x, const float* __restrict__ Wr1,
    const float* __restrict__ Wr2, float* __restrict__ gate,
    float* __restrict__ probs)
{
  __shared__ float red[64][RD + 1];
  __shared__ float lg[NEXP];
  __shared__ float pr[NEXP];
  const int t   = blockIdx.x;
  const int tid = threadIdx.x;
  const float* xp = x + (size_t)t * DDIM;

  float r[RD];
#pragma unroll
  for (int j = 0; j < RD; ++j) r[j] = 0.f;
  for (int d = tid; d < DDIM; d += 64) {
    float xv = xp[d];
    const float* w = Wr1 + d * RD;
#pragma unroll
    for (int j = 0; j < RD; ++j) r[j] += xv * w[j];
  }
#pragma unroll
  for (int j = 0; j < RD; ++j) red[tid][j] = r[j];
  __syncthreads();
  for (int s = 32; s > 0; s >>= 1) {
    if (tid < s) {
#pragma unroll
      for (int j = 0; j < RD; ++j) red[tid][j] += red[tid + s][j];
    }
    __syncthreads();
  }
  float lsum = 0.f;
#pragma unroll
  for (int j = 0; j < RD; ++j) lsum += red[0][j] * Wr2[j * NEXP + tid];
  lg[tid] = lsum;
  __syncthreads();
  float m = -1e30f;
  for (int i = 0; i < NEXP; ++i) m = fmaxf(m, lg[i]);
  float ssum = 0.f;
  for (int i = 0; i < NEXP; ++i) ssum += __expf(lg[i] - m);
  float p = __expf(lsum - m) / ssum;
  pr[tid] = p;
  __syncthreads();
  int c = 0;
  for (int i = 0; i < NEXP; ++i) {
    float q = pr[i];
    c += (q > p || (q == p && i < tid)) ? 1 : 0;
  }
  probs[(size_t)t * NEXP + tid] = p;
  gate [(size_t)t * NEXP + tid] = (c < KSEL) ? p : 0.f;
}

// ------------------------------------------------------------------
// WMMA fragment loaders (CDNA5 16-bit layouts, wave32)
// ------------------------------------------------------------------
__device__ inline v16bf load_frag_a(const __bf16* tile, int row, int hi) {
  const __bf16* p = tile + row * LDT + hi * 8;
  v8bf lo = *(const v8bf*)p;
  v8bf h2 = *(const v8bf*)(p + 16);
  return __builtin_shufflevector(lo, h2, 0,1,2,3,4,5,6,7,8,9,10,11,12,13,14,15);
}
__device__ inline v16bf load_frag_b(const __bf16* tile, int col, int hi) {
  const __bf16* p = tile + col * LDT + hi * 8;
  v8bf lo = *(const v8bf*)p;
  v8bf h2 = *(const v8bf*)(p + 16);
  return __builtin_shufflevector(lo, h2, 0,1,2,3,4,5,6,7,8,9,10,11,12,13,14,15);
}

// ------------------------------------------------------------------
// GEMM1: Zg[t,n] = (sum_k x[t,k]*A_w[n,k]) * gate[t, n/8]   (bf16 out)
// x tile: register-staged f32->bf16; weight tile: TDM, double-buffered
// ------------------------------------------------------------------
__global__ __launch_bounds__(256) void gemm_xa_gate(
    const float*  __restrict__ X,    // TOK x DDIM f32
    const __bf16* __restrict__ Aw,   // ERANK x DDIM bf16 (N-major)
    const float*  __restrict__ gate, // TOK x NEXP
    __bf16*       __restrict__ Zg)   // TOK x ERANK
{
  __shared__ __align__(16) __bf16 At[2][BM * LDT];
  __shared__ __align__(16) __bf16 Bt[2][BN * LDT];
  __shared__ float Gt[BM * 16];
  const int tid = threadIdx.x;
  const int rb = blockIdx.y * BM;
  const int cb = blockIdx.x * BN;
#pragma unroll
  for (int i = 0; i < (BM * 16) / 256; ++i) {
    int e = tid + i * 256;
    Gt[e] = gate[(size_t)(rb + (e >> 4)) * NEXP + (cb >> 3) + (e & 15)];
  }
  const int lane = tid & 31;
  // scalar (SGPR) wave id -> real s_cbranch, so exactly one wave issues TDM
  const int wv = __builtin_amdgcn_readfirstlane(tid >> 5);
  const int wave = tid >> 5;
  const int wm = wave & 1, wn = wave >> 1;     // 2 x 4 wave grid
  const int hi = lane >> 4, r16 = lane & 15;
  const int lrow = tid >> 2;                   // 0..63 (unit u adds +64)
  const int lch  = (tid & 3) * 8;

  v8f acc[4][2] = {};
  float4 fa[2][2];

  // ---- prologue: stage A(0) in regs, DMA B(0) into Bt[0] ----
#pragma unroll
  for (int u = 0; u < 2; ++u) {
    const float4* src = (const float4*)(X + (size_t)(rb + lrow + u * 64) * DDIM + lch);
    fa[u][0] = src[0]; fa[u][1] = src[1];
  }
  if (wv == 0)
    tdm_load_tile_bf16((unsigned)(size_t)&Bt[0][0],
                       Aw + (size_t)cb * DDIM, (unsigned long long)DDIM);

  const int NK = DDIM / BK;                    // 64
  for (int i = 0; i < NK; ++i) {
    const int cur = i & 1;
    // commit staged x regs -> LDS (bf16)
#pragma unroll
    for (int u = 0; u < 2; ++u) {
      v8bf h;
      h[0] = (__bf16)fa[u][0].x; h[1] = (__bf16)fa[u][0].y;
      h[2] = (__bf16)fa[u][0].z; h[3] = (__bf16)fa[u][0].w;
      h[4] = (__bf16)fa[u][1].x; h[5] = (__bf16)fa[u][1].y;
      h[6] = (__bf16)fa[u][1].z; h[7] = (__bf16)fa[u][1].w;
      *(v8bf*)(&At[cur][(lrow + u * 64) * LDT + lch]) = h;
    }
    if (wv == 0) __builtin_amdgcn_s_wait_tensorcnt((short)0);
    __syncthreads();                           // tile i fully in LDS
    // issue next tile: global loads + TDM overlap with compute below
    if (i + 1 < NK) {
      const int k0 = (i + 1) * BK;
#pragma unroll
      for (int u = 0; u < 2; ++u) {
        const float4* src = (const float4*)(X + (size_t)(rb + lrow + u * 64) * DDIM + k0 + lch);
        fa[u][0] = src[0]; fa[u][1] = src[1];
      }
      if (wv == 0)
        tdm_load_tile_bf16((unsigned)(size_t)&Bt[1 - cur][0],
                           Aw + (size_t)cb * DDIM + k0, (unsigned long long)DDIM);
    }
    // ---- 8 WMMAs on tile i ----
    v16bf af[4], bfg[2];
#pragma unroll
    for (int mi = 0; mi < 4; ++mi) af[mi]  = load_frag_a(At[cur], wm * 64 + mi * 16 + r16, hi);
#pragma unroll
    for (int ni = 0; ni < 2; ++ni) bfg[ni] = load_frag_b(Bt[cur], wn * 32 + ni * 16 + r16, hi);
#pragma unroll
    for (int mi = 0; mi < 4; ++mi)
#pragma unroll
      for (int ni = 0; ni < 2; ++ni)
        acc[mi][ni] = __builtin_amdgcn_wmma_f32_16x16x32_bf16(
            false, af[mi], false, bfg[ni], (short)0, acc[mi][ni], false, false);
    __syncthreads();                           // reads done before buffer reuse
  }
  // epilogue: gate + bf16 store
#pragma unroll
  for (int mi = 0; mi < 4; ++mi) {
    int lr0 = wm * 64 + mi * 16 + hi * 8;
#pragma unroll
    for (int ni = 0; ni < 2; ++ni) {
      int lc = wn * 32 + ni * 16 + r16;
      int ex = lc >> 3;
#pragma unroll
      for (int v = 0; v < 8; ++v) {
        int lr = lr0 + v;
        float g = Gt[lr * 16 + ex];
        Zg[(size_t)(rb + lr) * ERANK + cb + lc] = (__bf16)(acc[mi][ni][v] * g);
      }
    }
  }
}

// ------------------------------------------------------------------
// GEMM2: Out[t,o] = SCALE_OUT * sum_k Zg[t,k]*B_w[o,k]   (f32 out)
// both tiles staged by TDM (wave0 -> A, wave1 -> B), double-buffered
// ------------------------------------------------------------------
__global__ __launch_bounds__(256) void gemm_zb_out(
    const __bf16* __restrict__ Zg,   // TOK x ERANK bf16
    const __bf16* __restrict__ Bw,   // DDIM x ERANK bf16 (N-major)
    float*        __restrict__ Out)  // TOK x DDIM
{
  __shared__ __align__(16) __bf16 At[2][BM * LDT];
  __shared__ __align__(16) __bf16 Bt[2][BN * LDT];
  const int tid = threadIdx.x;
  const int rb = blockIdx.y * BM;
  const int cb = blockIdx.x * BN;
  const int lane = tid & 31;
  const int wv = __builtin_amdgcn_readfirstlane(tid >> 5);
  const int wave = tid >> 5;
  const int wm = wave & 1, wn = wave >> 1;
  const int hi = lane >> 4, r16 = lane & 15;

  v8f acc[4][2] = {};

  if (wv == 0)
    tdm_load_tile_bf16((unsigned)(size_t)&At[0][0],
                       Zg + (size_t)rb * ERANK, (unsigned long long)ERANK);
  if (wv == 1)
    tdm_load_tile_bf16((unsigned)(size_t)&Bt[0][0],
                       Bw + (size_t)cb * ERANK, (unsigned long long)ERANK);

  const int NK = ERANK / BK;                   // 16
  for (int i = 0; i < NK; ++i) {
    const int cur = i & 1;
    if (i + 1 < NK) {
      const int k0 = (i + 1) * BK;
      if (wv == 0)
        tdm_load_tile_bf16((unsigned)(size_t)&At[1 - cur][0],
                           Zg + (size_t)rb * ERANK + k0, (unsigned long long)ERANK);
      if (wv == 1)
        tdm_load_tile_bf16((unsigned)(size_t)&Bt[1 - cur][0],
                           Bw + (size_t)cb * ERANK + k0, (unsigned long long)ERANK);
      if (wv < 2) __builtin_amdgcn_s_wait_tensorcnt((short)1); // own tile(i) done
    } else {
      if (wv < 2) __builtin_amdgcn_s_wait_tensorcnt((short)0);
    }
    __syncthreads();
    v16bf af[4], bfg[2];
#pragma unroll
    for (int mi = 0; mi < 4; ++mi) af[mi]  = load_frag_a(At[cur], wm * 64 + mi * 16 + r16, hi);
#pragma unroll
    for (int ni = 0; ni < 2; ++ni) bfg[ni] = load_frag_b(Bt[cur], wn * 32 + ni * 16 + r16, hi);
#pragma unroll
    for (int mi = 0; mi < 4; ++mi)
#pragma unroll
      for (int ni = 0; ni < 2; ++ni)
        acc[mi][ni] = __builtin_amdgcn_wmma_f32_16x16x32_bf16(
            false, af[mi], false, bfg[ni], (short)0, acc[mi][ni], false, false);
    __syncthreads();
  }
#pragma unroll
  for (int mi = 0; mi < 4; ++mi) {
    int lr0 = wm * 64 + mi * 16 + hi * 8;
#pragma unroll
    for (int ni = 0; ni < 2; ++ni) {
      int lc = wn * 32 + ni * 16 + r16;
#pragma unroll
      for (int v = 0; v < 8; ++v) {
        int lr = lr0 + v;
        Out[(size_t)(rb + lr) * DDIM + cb + lc] = acc[mi][ni][v] * SCALE_OUT;
      }
    }
  }
}

// ------------------------------------------------------------------
// Aux loss:  aux = E * sum_e (cnt_e/T) * (psum_e/T)
// ------------------------------------------------------------------
__global__ __launch_bounds__(256) void aux_partial(
    const float* __restrict__ gate, const float* __restrict__ probs,
    float* __restrict__ partial)
{
  const int e = blockIdx.x;
  const int tid = threadIdx.x;
  float cnt = 0.f, ps = 0.f;
  for (int t = tid; t < TOK; t += 256) {
    cnt += (gate[(size_t)t * NEXP + e] != 0.f) ? 1.f : 0.f;
    ps  += probs[(size_t)t * NEXP + e];
  }
  __shared__ float sc[256], sp[256];
  sc[tid] = cnt; sp[tid] = ps;
  __syncthreads();
  for (int s = 128; s > 0; s >>= 1) {
    if (tid < s) { sc[tid] += sc[tid + s]; sp[tid] += sp[tid + s]; }
    __syncthreads();
  }
  if (tid == 0) {
    const float invT = 1.0f / (float)TOK;
    partial[e] = (float)NEXP * (sc[0] * invT) * (sp[0] * invT);
  }
}

__global__ __launch_bounds__(64) void aux_final(const float* __restrict__ partial,
                                                float* __restrict__ out_aux)
{
  __shared__ float s[64];
  const int tid = threadIdx.x;
  s[tid] = partial[tid];
  __syncthreads();
  for (int st = 32; st > 0; st >>= 1) {
    if (tid < st) s[tid] += s[tid + st];
    __syncthreads();
  }
  if (tid == 0) *out_aux = s[0];
}

// ------------------------------------------------------------------
extern "C" void kernel_launch(void* const* d_in, const int* in_sizes, int n_in,
                              void* d_out, int out_size, void* d_ws, size_t ws_size,
                              hipStream_t stream) {
  (void)in_sizes; (void)n_in; (void)out_size; (void)ws_size;
  const float* x   = (const float*)d_in[0];
  const float* A_w = (const float*)d_in[1];
  const float* B_w = (const float*)d_in[2];
  const float* Wr1 = (const float*)d_in[3];
  const float* Wr2 = (const float*)d_in[4];
  float* out = (float*)d_out;

  // workspace layout (bytes)
  char* ws = (char*)d_ws;
  const size_t off_probs = (size_t)TOK * NEXP * 4;                  // gate:  4 MB
  const size_t off_abf   = off_probs + (size_t)TOK * NEXP * 4;      // probs: 4 MB
  const size_t off_bbf   = off_abf   + (size_t)ERANK * DDIM * 2;    // Abf:   2 MB
  const size_t off_zg    = off_bbf   + (size_t)DDIM * ERANK * 2;    // Bbf:   2 MB
  const size_t off_part  = off_zg    + (size_t)TOK * ERANK * 2;     // Zg:   16 MB
  float*  gate  = (float*)(ws);
  float*  probs = (float*)(ws + off_probs);
  __bf16* Abf   = (__bf16*)(ws + off_abf);
  __bf16* Bbf   = (__bf16*)(ws + off_bbf);
  __bf16* Zg    = (__bf16*)(ws + off_zg);
  float*  part  = (float*)(ws + off_part);

  cvt_bf16<<<(ERANK * DDIM) / 256, 256, 0, stream>>>(A_w, Abf, ERANK * DDIM);
  cvt_bf16<<<(DDIM * ERANK) / 256, 256, 0, stream>>>(B_w, Bbf, DDIM * ERANK);
  router_kernel<<<TOK, 64, 0, stream>>>(x, Wr1, Wr2, gate, probs);
  gemm_xa_gate<<<dim3(ERANK / BN, TOK / BM), 256, 0, stream>>>(x, Abf, gate, Zg);
  gemm_zb_out<<<dim3(DDIM / BN, TOK / BM), 256, 0, stream>>>(Zg, Bbf, out);
  aux_partial<<<NEXP, 256, 0, stream>>>(gate, probs, part);
  aux_final<<<1, 64, 0, stream>>>(part, out + (size_t)TOK * DDIM);
}
